// Net_29746943492301
// MI455X (gfx1250) — matrix operand order repaired and checked by reading, not compiled, not run on hardware
//
#include <hip/hip_runtime.h>
#include <hip/hip_bf16.h>
#include <math.h>

// ---------------------------------------------------------------------------
// Problem constants (from reference)
// ---------------------------------------------------------------------------
#define N0c  262144
#define N1c  32768
#define N2c  4096
#define E1c  327680
#define E2c  40960
#define FINc 512
#define Hc   256
#define OUTc 128
#define HCc  128
#define Cc   40
#define CPc  48   // classes padded to multiple of 16 for WMMA N-tiles

typedef __attribute__((ext_vector_type(16))) __bf16 v16bf;
typedef __attribute__((ext_vector_type(8)))  float  v8f;

// f32 -> bf16 round-to-nearest-even via bit ops (no reliance on native cvt)
__device__ __forceinline__ __bf16 f2bf(float f) {
    unsigned int u = __builtin_bit_cast(unsigned int, f);
    unsigned int r = (u + 0x7FFFu + ((u >> 16) & 1u)) >> 16;
    return __builtin_bit_cast(__bf16, (unsigned short)r);
}

// ---------------------------------------------------------------------------
// Zero-fill (grid-stride)
// ---------------------------------------------------------------------------
__global__ void zero_f32(float* __restrict__ p, long long n) {
    long long i = (long long)blockIdx.x * blockDim.x + threadIdx.x;
    long long st = (long long)gridDim.x * blockDim.x;
    for (; i < n; i += st) p[i] = 0.0f;
}

// ---------------------------------------------------------------------------
// Edge-parallel scatter-add (segment-sum) + degree count.
// One thread handles 4 features of one edge; atomics land in L2 (agg fits).
// ---------------------------------------------------------------------------
__global__ __launch_bounds__(256) void scatter_mean_accum(
    const float* __restrict__ feat, const long long* __restrict__ src,
    const long long* __restrict__ dst, float* __restrict__ agg,
    float* __restrict__ cnt, int E, int F)
{
    const int tpe = F >> 2; // threads per edge
    long long gid  = (long long)blockIdx.x * blockDim.x + threadIdx.x;
    long long edge = gid / tpe;
    int       part = (int)(gid - edge * tpe);
    if (edge >= E) return;
    long long s = src[edge];
    long long d = dst[edge];
    const float4 v = *(const float4*)(feat + s * (long long)F + part * 4);
    float* o = agg + d * (long long)F + part * 4;
    atomicAdd(o + 0, v.x);
    atomicAdd(o + 1, v.y);
    atomicAdd(o + 2, v.z);
    atomicAdd(o + 3, v.w);
    if (part == 0) atomicAdd(cnt + d, 1.0f);
}

// ---------------------------------------------------------------------------
// Build concatenated activation matrix:  xcat[n] = [ agg[n]/max(cnt,1) , xdst[n] ]
// (bf16 output, row length 2F)
// ---------------------------------------------------------------------------
__global__ __launch_bounds__(256) void build_xcat(
    const float* __restrict__ agg, const float* __restrict__ cnt,
    const float* __restrict__ xd, __bf16* __restrict__ xcat, int N, int F)
{
    long long i = (long long)blockIdx.x * blockDim.x + threadIdx.x;
    if (i >= (long long)N * F) return;
    int n = (int)(i / F);
    int f = (int)(i - (long long)n * F);
    float c = cnt[n]; c = c > 1.0f ? c : 1.0f;
    __bf16* row = xcat + (size_t)n * (2 * F);
    row[f]     = f2bf(agg[i] * (1.0f / c));
    row[F + f] = f2bf(xd[i]);
}

// ---------------------------------------------------------------------------
// Pack TRANSPOSED vstack(W_l, W_r) -> bf16.  Output Wt[N][2K] (row-major),
// i.e. Wt[j][k] = (k<K ? Wl[k][j] : Wr[k-K][j]).  Transposed so the GEMM's
// B-fragment (one column, two runs of 8 consecutive K) is two contiguous b128 loads.
// ---------------------------------------------------------------------------
__global__ __launch_bounds__(256) void build_wcat_t(
    const float* __restrict__ Wl, const float* __restrict__ Wr,
    __bf16* __restrict__ Wt, int K, int N)
{
    long long i = (long long)blockIdx.x * blockDim.x + threadIdx.x;
    if (i >= (long long)N * 2 * K) return;
    long long j = i / (2 * K);          // output column of original B
    long long k = i - j * (2 * K);      // K index
    float v = (k < K) ? Wl[k * N + j] : Wr[(k - K) * N + j];
    Wt[i] = f2bf(v);
}

// Transpose single weight matrix W[K][N] -> Wt[N][K] bf16
__global__ __launch_bounds__(256) void transpose_w(
    const float* __restrict__ W, __bf16* __restrict__ Wt, int K, int N)
{
    long long i = (long long)blockIdx.x * blockDim.x + threadIdx.x;
    if (i >= (long long)N * K) return;
    long long j = i / K, k = i - j * K;
    Wt[i] = f2bf(W[k * N + j]);
}

// Pad + transpose classifier head: Wc2[128][40] -> Wt[48][128] bf16, bc2 -> [48]
__global__ __launch_bounds__(256) void pad_head_t(
    const float* __restrict__ W, const float* __restrict__ b,
    __bf16* __restrict__ Wt, float* __restrict__ bp)
{
    int i = blockIdx.x * blockDim.x + threadIdx.x;
    if (i < CPc * HCc) {
        int j = i / HCc, k = i - j * HCc;   // j = class col, k = K index
        Wt[i] = f2bf(j < Cc ? W[k * Cc + j] : 0.0f);
    }
    if (i < CPc) bp[i] = (i < Cc) ? b[i] : 0.0f;
}

// ---------------------------------------------------------------------------
// Fragment loaders (CDNA5 ISA 7.12.2 layouts, wave32):
//   A 16x32: lane m=lane&15 holds row m; hi=lane>>4;
//            elems 0..7 -> K = 8*hi+0..7, elems 8..15 -> K = 8*hi+16..23
//            => two contiguous 16B chunks at Arow + k0 (+16).
//   B 32x16: lane col = lane&15; same K mapping as A:
//            elems 0..7 -> K = 8*hi+0..7, elems 8..15 -> K = 8*hi+16..23
//            => with Bt[N][K], two 16B chunks at Brow + k0 (+16).
//   C/D f32: vgpr r -> row r + 8*hi, col lane&15.
// ---------------------------------------------------------------------------
__device__ __forceinline__ v16bf load_a_frag(const __bf16* p) {
    union { uint4 u; __bf16 h[8]; } cl, ch;
    cl.u = *(const uint4*)(p);
    ch.u = *(const uint4*)(p + 16);
    v16bf a;
#pragma unroll
    for (int i = 0; i < 8; ++i) { a[i] = cl.h[i]; a[i + 8] = ch.h[i]; }
    return a;
}
__device__ __forceinline__ v16bf load_b_frag(const __bf16* p) {
    union { uint4 u; __bf16 h[8]; } bl, bh;
    bl.u = *(const uint4*)(p);
    bh.u = *(const uint4*)(p + 16);
    v16bf b;
#pragma unroll
    for (int i = 0; i < 8; ++i) { b[i] = bl.h[i]; b[i + 8] = bh.h[i]; }
    return b;
}

// ---------------------------------------------------------------------------
// bf16 WMMA GEMM:  out = act( A[MxK]_bf16 @ B[KxN]_bf16 + bias[N] )
// B supplied TRANSPOSED as Bt[N][K].  Requires K % 64 == 0.
// Block = 128 threads (4 waves). Wave tile = 16 rows x (16*WN) cols:
// A fragment loaded once per k-step and reused across WN column tiles.
// K-loop is PING-PONG double-buffered with unroll-by-2: each buffer set is
// loaded and consumed in place (no register rotation copies); loads for the
// next 32-K slab overlap the WMMAs of the current one (partial s_wait_loadcnt).
// ---------------------------------------------------------------------------
template <int WN>
__global__ __launch_bounds__(128) void gemm_bf16_wmma(
    const __bf16* __restrict__ A, const __bf16* __restrict__ Bt,
    const float* __restrict__ bias,
    float* __restrict__ outF, __bf16* __restrict__ outB,
    int M, int N, int K, int relu)
{
    const int lane = threadIdx.x & 31;
    const int wave = threadIdx.x >> 5;
    const int col0 = blockIdx.x * (16 * WN);
    const int row0 = blockIdx.y * 64 + wave * 16;
    if (row0 >= M) return; // wave-uniform

    const int m  = lane & 15;
    const int hi = lane >> 4;

    v8f acc[WN];
#pragma unroll
    for (int t = 0; t < WN; ++t) {
        const float bv = bias[col0 + t * 16 + m];
#pragma unroll
        for (int r = 0; r < 8; ++r) acc[t][r] = bv;
    }

    const __bf16* Arow = A + (size_t)(row0 + m) * K + hi * 8;
    const __bf16* Brow[WN];
#pragma unroll
    for (int t = 0; t < WN; ++t)
        Brow[t] = Bt + (size_t)(col0 + t * 16 + m) * K + hi * 8;

    // --- prologue: buffer 0 <- k=0 ---
    v16bf a0 = load_a_frag(Arow);
    v16bf b0[WN];
#pragma unroll
    for (int t = 0; t < WN; ++t) b0[t] = load_b_frag(Brow[t]);
    v16bf a1;
    v16bf b1[WN];

    // --- steady state: ping-pong, 64 K per iteration ---
    int k0 = 0;
    for (; k0 < K - 64; k0 += 64) {
        a1 = load_a_frag(Arow + k0 + 32);
#pragma unroll
        for (int t = 0; t < WN; ++t) b1[t] = load_b_frag(Brow[t] + k0 + 32);
#pragma unroll
        for (int t = 0; t < WN; ++t)
            acc[t] = __builtin_amdgcn_wmma_f32_16x16x32_bf16(
                false, a0, false, b0[t], (short)0, acc[t], false, false);

        a0 = load_a_frag(Arow + k0 + 64);
#pragma unroll
        for (int t = 0; t < WN; ++t) b0[t] = load_b_frag(Brow[t] + k0 + 64);
#pragma unroll
        for (int t = 0; t < WN; ++t)
            acc[t] = __builtin_amdgcn_wmma_f32_16x16x32_bf16(
                false, a1, false, b1[t], (short)0, acc[t], false, false);
    }
    // --- tail: buf0 holds K-64, load buf1 <- K-32 ---
    a1 = load_a_frag(Arow + K - 32);
#pragma unroll
    for (int t = 0; t < WN; ++t) b1[t] = load_b_frag(Brow[t] + K - 32);
#pragma unroll
    for (int t = 0; t < WN; ++t)
        acc[t] = __builtin_amdgcn_wmma_f32_16x16x32_bf16(
            false, a0, false, b0[t], (short)0, acc[t], false, false);
#pragma unroll
    for (int t = 0; t < WN; ++t)
        acc[t] = __builtin_amdgcn_wmma_f32_16x16x32_bf16(
            false, a1, false, b1[t], (short)0, acc[t], false, false);

#pragma unroll
    for (int t = 0; t < WN; ++t) {
#pragma unroll
        for (int r = 0; r < 8; ++r) {
            int row = row0 + r + hi * 8;
            float v = acc[t][r];
            if (relu) v = fmaxf(v, 0.0f);
            size_t idx = (size_t)row * N + col0 + t * 16 + m;
            if (outF) outF[idx] = v;
            if (outB) outB[idx] = f2bf(v);
        }
    }
}

// ---------------------------------------------------------------------------
// Row-wise log-softmax over first Cc of CPc padded cols; one wave per row.
// ---------------------------------------------------------------------------
__global__ __launch_bounds__(256) void log_softmax_rows(
    const float* __restrict__ h, float* __restrict__ out, int M)
{
    int wave = threadIdx.x >> 5, lane = threadIdx.x & 31;
    int row = blockIdx.x * 8 + wave;
    if (row >= M) return;
    const float* r = h + (size_t)row * CPc;
    const float NEG = -__builtin_inff();
    float v0 = (lane < Cc) ? r[lane] : NEG;
    float v1 = (lane + 32 < Cc) ? r[lane + 32] : NEG;
    float mx = fmaxf(v0, v1);
#pragma unroll
    for (int o = 16; o >= 1; o >>= 1) mx = fmaxf(mx, __shfl_xor(mx, o, 32));
    float s = 0.0f;
    if (lane < Cc)      s += expf(v0 - mx);
    if (lane + 32 < Cc) s += expf(v1 - mx);
#pragma unroll
    for (int o = 16; o >= 1; o >>= 1) s += __shfl_xor(s, o, 32);
    float ls = logf(s);
    if (lane < Cc)      out[(size_t)row * Cc + lane]      = v0 - mx - ls;
    if (lane + 32 < Cc) out[(size_t)row * Cc + lane + 32] = v1 - mx - ls;
}

// ---------------------------------------------------------------------------
// Host-side orchestration
// ---------------------------------------------------------------------------
extern "C" void kernel_launch(void* const* d_in, const int* in_sizes, int n_in,
                              void* d_out, int out_size, void* d_ws, size_t ws_size,
                              hipStream_t stream) {
    (void)in_sizes; (void)n_in; (void)out_size; (void)ws_size;

    const float*     x   = (const float*)d_in[0];
    const long long* e1s = (const long long*)d_in[1];
    const long long* e1d = (const long long*)d_in[2];
    const long long* e2s = (const long long*)d_in[3];
    const long long* e2d = (const long long*)d_in[4];
    const float* Wl1 = (const float*)d_in[5];
    const float* Wr1 = (const float*)d_in[6];
    const float* b1  = (const float*)d_in[7];
    const float* Wl2 = (const float*)d_in[8];
    const float* Wr2 = (const float*)d_in[9];
    const float* b2  = (const float*)d_in[10];
    const float* Wc1 = (const float*)d_in[11];
    const float* bc1 = (const float*)d_in[12];
    const float* Wc2 = (const float*)d_in[13];
    const float* bc2 = (const float*)d_in[14];
    float* out = (float*)d_out;

    // --- workspace carve-up (256B aligned) ---
    char* w = (char*)d_ws;
    auto carve = [&](size_t bytes) {
        void* p = (void*)w;
        w += (bytes + 255) & ~(size_t)255;
        return p;
    };
    // agg1,cnt1,agg2,cnt2 contiguous (all 256B multiples) -> single zero pass
    float*  agg1  = (float*)carve((size_t)N1c * FINc * 4);
    float*  cnt1  = (float*)carve((size_t)N1c * 4);
    float*  agg2  = (float*)carve((size_t)N2c * Hc * 4);
    float*  cnt2  = (float*)carve((size_t)N2c * 4);
    __bf16* xcat1 = (__bf16*)carve((size_t)N1c * (2 * FINc) * 2);
    __bf16* wcat1t = (__bf16*)carve((size_t)Hc * (2 * FINc) * 2);   // [256][1024]
    float*  h1    = (float*)carve((size_t)N1c * Hc * 4);
    __bf16* xcat2 = (__bf16*)carve((size_t)N2c * (2 * Hc) * 2);
    __bf16* wcat2t = (__bf16*)carve((size_t)OUTc * (2 * Hc) * 2);   // [128][512]
    __bf16* h2b   = (__bf16*)carve((size_t)N2c * OUTc * 2);
    __bf16* wc1t  = (__bf16*)carve((size_t)HCc * OUTc * 2);         // [128][128]
    __bf16* h3b   = (__bf16*)carve((size_t)N2c * HCc * 2);
    __bf16* wc2t  = (__bf16*)carve((size_t)CPc * HCc * 2);          // [48][128]
    float*  bc2p  = (float*)carve((size_t)CPc * 4);
    float*  h4    = (float*)carve((size_t)N2c * CPc * 4);

    const long long zeroN =
        ((long long)N1c * FINc + N1c + (long long)N2c * Hc + N2c);

    // 1) zero accumulators
    zero_f32<<<4096, 256, 0, stream>>>(agg1, zeroN);

    // 2) layer-1 scatter (E1 edges x 512 feats, float4/thread)
    {
        long long threads = (long long)E1c * (FINc >> 2);
        int blocks = (int)((threads + 255) / 256);
        scatter_mean_accum<<<blocks, 256, 0, stream>>>(x, e1s, e1d, agg1, cnt1, E1c, FINc);
    }
    // 3) xcat1 = [agg1/cnt1 , x[:N1]]  (bf16)
    {
        long long n = (long long)N1c * FINc;
        build_xcat<<<(int)((n + 255) / 256), 256, 0, stream>>>(agg1, cnt1, x, xcat1, N1c, FINc);
    }
    // 4) wcat1t = vstack(W_l1, W_r1)^T (bf16, [256][1024])
    {
        long long n = (long long)Hc * 2 * FINc;
        build_wcat_t<<<(int)((n + 255) / 256), 256, 0, stream>>>(Wl1, Wr1, wcat1t, FINc, Hc);
    }
    // 5) h1 = relu(xcat1 @ wcat1 + b1)   [32768x1024]@[1024x256]
    gemm_bf16_wmma<4><<<dim3(Hc / 64, N1c / 64), 128, 0, stream>>>(
        xcat1, wcat1t, b1, h1, nullptr, N1c, Hc, 2 * FINc, 1);

    // 6) layer-2 scatter over h1
    {
        long long threads = (long long)E2c * (Hc >> 2);
        int blocks = (int)((threads + 255) / 256);
        scatter_mean_accum<<<blocks, 256, 0, stream>>>(h1, e2s, e2d, agg2, cnt2, E2c, Hc);
    }
    // 7) xcat2 = [agg2/cnt2 , h1[:N2]]
    {
        long long n = (long long)N2c * Hc;
        build_xcat<<<(int)((n + 255) / 256), 256, 0, stream>>>(agg2, cnt2, h1, xcat2, N2c, Hc);
    }
    // 8) wcat2t = vstack(W_l2, W_r2)^T  ([128][512])
    {
        long long n = (long long)OUTc * 2 * Hc;
        build_wcat_t<<<(int)((n + 255) / 256), 256, 0, stream>>>(Wl2, Wr2, wcat2t, Hc, OUTc);
    }
    // 9) h2 = xcat2 @ wcat2 + b2 (no relu)  -> bf16 only
    gemm_bf16_wmma<4><<<dim3(OUTc / 64, N2c / 64), 128, 0, stream>>>(
        xcat2, wcat2t, b2, nullptr, h2b, N2c, OUTc, 2 * Hc, 0);

    // 10) classifier weights -> bf16 transposed
    {
        long long n = (long long)HCc * OUTc;
        transpose_w<<<(int)((n + 255) / 256), 256, 0, stream>>>(Wc1, wc1t, OUTc, HCc);
    }
    // 11) h3 = relu(h2 @ Wc1 + bc1) -> bf16
    gemm_bf16_wmma<4><<<dim3(HCc / 64, N2c / 64), 128, 0, stream>>>(
        h2b, wc1t, bc1, nullptr, h3b, N2c, HCc, OUTc, 1);

    // 12) pad + transpose head to [48][128]
    pad_head_t<<<(CPc * HCc + 255) / 256, 256, 0, stream>>>(Wc2, bc2, wc2t, bc2p);
    // 13) h4 = relu(h3 @ Wc2p + bc2p)  [4096x128]@[128x48]
    gemm_bf16_wmma<3><<<dim3(1, N2c / 64), 128, 0, stream>>>(
        h3b, wc2t, bc2p, h4, nullptr, N2c, CPc, HCc, 1);

    // 14) log_softmax over first 40 cols
    log_softmax_rows<<<(N2c + 7) / 8, 256, 0, stream>>>(h4, out, N2c);
}
